// LSTM_73100343378117
// MI455X (gfx1250) — matrix operand order repaired and checked by reading, not compiled, not run on hardware
//
#include <hip/hip_runtime.h>
#include <hip/hip_bf16.h>

// LSTM on MI455X (gfx1250): bf16 WMMA recurrence, fused cell + next-step
// prologue epilogue, ping-pong bf16 activation buffers.
#define TB   2048      // trajectories (batch)
#define TT   32        // rollout
#define IDIM 1024      // input dim
#define HDIM 1024      // hidden dim
#define GDIM 4096      // 4*HDIM

typedef unsigned short ushort_t;
typedef __attribute__((ext_vector_type(16))) __bf16 v16bf;
typedef __attribute__((ext_vector_type(8)))  __bf16 v8bf;
typedef __attribute__((ext_vector_type(8)))  float  v8f;

// ---- helpers ---------------------------------------------------------------

__device__ __forceinline__ ushort_t f2bf(float f) {
    unsigned int u = __float_as_uint(f);          // RNE fp32 -> bf16
    u += 0x7FFFu + ((u >> 16) & 1u);
    return (ushort_t)(u >> 16);
}

__device__ __forceinline__ float sigmoidf_fast(float x) {
    return 1.0f / (1.0f + __expf(-x));
}

// ---- one-time setup --------------------------------------------------------

__global__ void pack_weights_kernel(const float* __restrict__ wih,
                                    const float* __restrict__ whh,
                                    const float* __restrict__ bih,
                                    const float* __restrict__ bhh,
                                    ushort_t* __restrict__ wihb,
                                    ushort_t* __restrict__ whhb,
                                    float* __restrict__ bias) {
    const int idx = blockIdx.x * 256 + threadIdx.x;       // 0 .. 4096*1024-1
    wihb[idx] = f2bf(wih[idx]);
    whhb[idx] = f2bf(whh[idx]);
    if (idx < GDIM) bias[idx] = bih[idx] + bhh[idx];
}

// c0 from rnn_states; bf16 h0 (t=0 reset is defined 0) and bf16 x_0.
__global__ void init_state_kernel(const float* __restrict__ rnn,
                                  const float* __restrict__ head,
                                  float* __restrict__ c,
                                  ushort_t* __restrict__ hbf0,
                                  ushort_t* __restrict__ xbf0) {
    const int idx = blockIdx.x * 256 + threadIdx.x;       // 0 .. 2048*1024-1
    const int b = idx >> 10, k = idx & 1023;
    c[idx]    = rnn[b * (2 * HDIM) + HDIM + k];
    hbf0[idx] = f2bf(rnn[b * (2 * HDIM) + k]);
    xbf0[idx] = f2bf(head[((size_t)b * TT + 0) * IDIM + k]);
}

// ---- fused WMMA GEMM + LSTM cell + next-step prologue ---------------------
// Workgroup: 256 threads = 8 waves. Tile = 128 batch rows x 32 hidden units.
// Each wave owns 32 rows x 16 units and accumulates all FOUR gates (8 accs),
// so each B (weight) fragment feeds 2 WMMAs. Epilogue performs the cell
// update and directly emits the bf16 (reset-masked) h and bf16 x for step
// t+1 into the ping-pong buffers — no separate prestep kernel.

__global__ __launch_bounds__(256)
void lstm_step_kernel(const ushort_t* __restrict__ xbf_r,
                      const ushort_t* __restrict__ hbf_r,
                      const ushort_t* __restrict__ wihb,
                      const ushort_t* __restrict__ whhb,
                      const float* __restrict__ bias,
                      const float* __restrict__ dones,
                      const float* __restrict__ head,
                      float* __restrict__ c,
                      ushort_t* __restrict__ hbf_w,
                      ushort_t* __restrict__ xbf_w,
                      float* __restrict__ xout,
                      int t) {
    const int tid   = threadIdx.x;
    const int wave  = tid >> 5;
    const int lane  = tid & 31;
    const int m_sub = wave >> 1;           // 0..3  (32-row subtile)
    const int n_sub = wave & 1;            // 0..1  (16-unit subtile)
    const int lhalf = lane >> 4;           // 0/1: K-split lane half
    const int lmod  = lane & 15;

    const int row_base  = blockIdx.y * 128 + m_sub * 32;  // batch rows
    const int unit_base = blockIdx.x * 32 + n_sub * 16;   // hidden units

    v8f acc[2][4] = {};                    // [row group][gate i,f,g,o]

    const int bunit = unit_base + lmod;    // B-fragment column (= weight row)

    #pragma unroll
    for (int phase = 0; phase < 2; ++phase) {
        // phase 0: x @ W_ih^T   phase 1: h @ W_hh^T   (K = 1024 each)
        const __bf16* Ab = (const __bf16*)(phase ? hbf_r : xbf_r);
        const __bf16* A0 = Ab + (size_t)(row_base + lmod) * 1024;
        const __bf16* A1 = A0 + (size_t)16 * 1024;
        const __bf16* W  = (const __bf16*)(phase ? whhb : wihb);
        const __bf16* Bg[4] = {
            W + ((size_t)(0 * HDIM + bunit)) * 1024,      // gate i
            W + ((size_t)(1 * HDIM + bunit)) * 1024,      // gate f
            W + ((size_t)(2 * HDIM + bunit)) * 1024,      // gate g
            W + ((size_t)(3 * HDIM + bunit)) * 1024 };    // gate o

        for (int k0 = 0; k0 < 1024; k0 += 32) {
            // A fragments (16x32 bf16): lanes 0-15 hold K [0..15],
            // lanes 16-31 hold K [16..31] of the chunk.
            v8bf a0lo = *(const v8bf*)(A0 + k0 + 16 * lhalf);
            v8bf a0hi = *(const v8bf*)(A0 + k0 + 8 + 16 * lhalf);
            v8bf a1lo = *(const v8bf*)(A1 + k0 + 16 * lhalf);
            v8bf a1hi = *(const v8bf*)(A1 + k0 + 8 + 16 * lhalf);
            v16bf a0 = __builtin_shufflevector(a0lo, a0hi,
                0, 1, 2, 3, 4, 5, 6, 7, 8, 9, 10, 11, 12, 13, 14, 15);
            v16bf a1 = __builtin_shufflevector(a1lo, a1hi,
                0, 1, 2, 3, 4, 5, 6, 7, 8, 9, 10, 11, 12, 13, 14, 15);

            // B fragments: each lane reads 32 contiguous bytes of one
            // weight row; lane half selects K 0-15 / 16-31.
            const int boff = k0 + 16 * lhalf;
            v16bf b[4];
            #pragma unroll
            for (int q = 0; q < 4; ++q)
                b[q] = *(const v16bf*)(Bg[q] + boff);

            __builtin_prefetch(A0 + k0 + 32, 0, 3);       // next K chunk

            #pragma unroll
            for (int q = 0; q < 4; ++q) {
                acc[0][q] = __builtin_amdgcn_wmma_f32_16x16x32_bf16(
                    false, a0, false, b[q], (short)0, acc[0][q], false, false);
                acc[1][q] = __builtin_amdgcn_wmma_f32_16x16x32_bf16(
                    false, a1, false, b[q], (short)0, acc[1][q], false, false);
            }
        }
    }

    // ---- fused LSTM cell + next-step prologue -----------------------------
    const int unit = unit_base + lmod;     // C layout: N = lane % 16
    const float bi = bias[0 * HDIM + unit];
    const float bf_ = bias[1 * HDIM + unit];
    const float bg = bias[2 * HDIM + unit];
    const float bo = bias[3 * HDIM + unit];

    #pragma unroll
    for (int rg = 0; rg < 2; ++rg) {
        #pragma unroll
        for (int r = 0; r < 8; ++r) {
            const int row = row_base + rg * 16 + r + 8 * lhalf;   // batch idx
            float gi = sigmoidf_fast(acc[rg][0][r] + bi);
            float gf = sigmoidf_fast(acc[rg][1][r] + bf_);
            float gg = tanhf(acc[rg][2][r] + bg);
            float go = sigmoidf_fast(acc[rg][3][r] + bo);

            float keep = 1.0f;                        // reset for THIS step
            if (t > 0) keep = 1.0f - dones[row * TT + (t - 1)];

            const size_t ci = (size_t)row * HDIM + unit;
            float cn = gf * (c[ci] * keep) + gi * gg;
            c[ci] = cn;
            float hn = go * tanhf(cn);

            // x_out[b, t, :] flat = (b*T + t)*H + unit
            xout[((size_t)row * TT + t) * HDIM + unit] = hn;

            if (t + 1 < TT) {
                // prologue for step t+1: reset-masked bf16 h, bf16 x_{t+1}
                float keepn = 1.0f - dones[row * TT + t];
                hbf_w[ci] = f2bf(hn * keepn);
                xbf_w[ci] = f2bf(head[((size_t)row * TT + (t + 1)) * IDIM + unit]);
            }
        }
    }
}

// ---- final: new_rnn_states = concat(h_final, c) ---------------------------
// h_final is x_out[:, T-1, :] — no separate h buffer needed.

__global__ void finalize_kernel(const float* __restrict__ xout,
                                const float* __restrict__ c,
                                float* __restrict__ out2) {
    const int idx = blockIdx.x * 256 + threadIdx.x;       // 0 .. 2048*1024-1
    const int b = idx >> 10, k = idx & 1023;
    out2[(size_t)b * (2 * HDIM) + k] =
        xout[((size_t)b * TT + (TT - 1)) * HDIM + k];
    out2[(size_t)b * (2 * HDIM) + HDIM + k] = c[idx];
}

// ---- host launcher ---------------------------------------------------------

extern "C" void kernel_launch(void* const* d_in, const int* in_sizes, int n_in,
                              void* d_out, int out_size, void* d_ws, size_t ws_size,
                              hipStream_t stream) {
    const float* head  = (const float*)d_in[0];   // (B*T, I)
    const float* rnn   = (const float*)d_in[1];   // (B, 2H)
    const float* dones = (const float*)d_in[2];   // (B*T,)
    const float* wih   = (const float*)d_in[3];   // (4H, I)
    const float* whh   = (const float*)d_in[4];   // (4H, H)
    const float* bih   = (const float*)d_in[5];   // (4H,)
    const float* bhh   = (const float*)d_in[6];   // (4H,)
    float* out = (float*)d_out;

    // workspace layout (~41 MB)
    char* ws = (char*)d_ws;
    size_t o = 0;
    ushort_t* wihb = (ushort_t*)(ws + o); o += (size_t)GDIM * IDIM * 2;   //  8 MB
    ushort_t* whhb = (ushort_t*)(ws + o); o += (size_t)GDIM * HDIM * 2;   //  8 MB
    float*    bias = (float*)(ws + o);    o += (size_t)GDIM * 4;          // 16 KB
    float*    cbuf = (float*)(ws + o);    o += (size_t)TB * HDIM * 4;     //  8 MB
    ushort_t* hbf[2], *xbf[2];
    hbf[0] = (ushort_t*)(ws + o); o += (size_t)TB * HDIM * 2;             //  4 MB
    hbf[1] = (ushort_t*)(ws + o); o += (size_t)TB * HDIM * 2;             //  4 MB
    xbf[0] = (ushort_t*)(ws + o); o += (size_t)TB * IDIM * 2;             //  4 MB
    xbf[1] = (ushort_t*)(ws + o); o += (size_t)TB * IDIM * 2;             //  4 MB
    (void)ws_size; (void)in_sizes; (void)n_in; (void)out_size;

    pack_weights_kernel<<<(GDIM * IDIM) / 256, 256, 0, stream>>>(
        wih, whh, bih, bhh, wihb, whhb, bias);
    init_state_kernel<<<(TB * HDIM) / 256, 256, 0, stream>>>(
        rnn, head, cbuf, hbf[0], xbf[0]);

    for (int t = 0; t < TT; ++t) {
        const int rb = t & 1, wb = (t + 1) & 1;       // ping-pong
        dim3 grid(HDIM / 32, TB / 128);               // 32 x 16 workgroups
        lstm_step_kernel<<<grid, 256, 0, stream>>>(
            xbf[rb], hbf[rb], wihb, whhb, bias, dones, head,
            cbuf, hbf[wb], xbf[wb], out, t);
    }

    finalize_kernel<<<(TB * HDIM) / 256, 256, 0, stream>>>(
        out, cbuf, out + (size_t)TB * TT * HDIM);
}
